// LlamaAttention_7232724927153
// MI455X (gfx1250) — matrix operand order repaired and checked by reading, not compiled, not run on hardware
//
#include <hip/hip_runtime.h>
#include <hip/hip_bf16.h>
#include <stdint.h>

// ---------------- types ----------------
typedef __bf16 bf16_t;
typedef __attribute__((ext_vector_type(16))) __bf16 v16bf;
typedef __attribute__((ext_vector_type(8)))  float  v8f;

// ---------------- problem constants ----------------
#define S_LEN 2048
#define D_DIM 2048
#define NH    32
#define NKVH  8
#define HD    64
#define NQKV  3072         // (32 + 8 + 8) * 64
#define ATT_SCALE 0.125f   // 1/sqrt(64)

// ---------------- workspace layout (bytes) ----------------
static const size_t OFF_HID  = 0;              // bf16 [2048][2048]  8 MiB
static const size_t OFF_WCAT = 8ull  << 20;    // bf16 [3072][2048] 12 MiB
static const size_t OFF_WO   = 20ull << 20;    // bf16 [2048][2048]  8 MiB
static const size_t OFF_QKV  = 28ull << 20;    // f32  [2048][3072] 24 MiB (reused as attn bf16)
static const size_t OFF_QH   = 52ull << 20;    // bf16 [32][2048][64] 8 MiB
static const size_t OFF_KH   = 60ull << 20;    // bf16 [8][2048][64]  2 MiB
static const size_t OFF_VT   = 62ull << 20;    // bf16 [8][64][2048]  2 MiB -> 64 MiB total

// ---------------- helpers ----------------
__global__ __launch_bounds__(256)
void cast_f32_bf16(const float* __restrict__ src, bf16_t* __restrict__ dst, int n) {
    int i = blockIdx.x * 256 + threadIdx.x;
    if (i < n) dst[i] = (bf16_t)src[i];
}

// ---------------- NT GEMM: C[M][N] = A[M][K] * B[N][K]^T, bf16 in / f32 out ----
// Block = 128 threads (4 waves). Each wave computes a 16x64 strip of C.
// B slab [64 cols][64 K] is staged in LDS with GLOBAL_LOAD_ASYNC_TO_LDS_B128,
// double-buffered and tracked with ASYNCcnt (4 async ops per wave per slab).
// A fragment (16x32 bf16, ISA 7.12.2): lane l -> row l%16, 16 contiguous K at (l/16)*16 (this code's order, same as B).
// B fragment (32x16 bf16):             lane l -> col l%16, 16 contiguous K at (l/16)*16.
__global__ __launch_bounds__(128)
void gemm_nt_wmma(const bf16_t* __restrict__ A, const bf16_t* __restrict__ B,
                  float* __restrict__ C, int M, int N, int K) {
    __shared__ __align__(16) bf16_t sB[2][64][64];   // 16 KiB, double buffered

    const int lane = threadIdx.x & 31;
    const int wave = threadIdx.x >> 5;
    const int lo = lane & 15;
    const int hi = lane >> 4;

    const int row0 = blockIdx.x * 64 + wave * 16;
    const int col0 = blockIdx.y * 64;

    const bf16_t* pa = A + (size_t)(row0 + lo) * K;

    // issue one 64x64 bf16 B slab (8 KiB) as 4 async b128 ops per wave
    auto issue_slab = [&](int kb, int bufIdx) {
#pragma unroll
        for (int i = 0; i < 4; ++i) {
            const int id  = i * 128 + threadIdx.x;   // 0..511 chunk id
            const int row = id >> 3;                 // 64 B-rows (columns of C)
            const int c16 = id & 7;                  // 8 chunks of 8 halves
            const bf16_t* g = B + (size_t)(col0 + row) * K + kb * 64 + c16 * 8;
            const unsigned lds = (unsigned)(uintptr_t)&sB[bufIdx][row][c16 * 8];
            asm volatile("global_load_async_to_lds_b128 %0, %1, off"
                         :: "v"(lds), "v"(g) : "memory");
        }
    };

    const v8f vzero = {0.f,0.f,0.f,0.f,0.f,0.f,0.f,0.f};
    v8f acc[4] = {vzero, vzero, vzero, vzero};

    const int nkb = K >> 6;        // K-slabs of 64
    issue_slab(0, 0);

    for (int kb = 0; kb < nkb; ++kb) {
        const int buf = kb & 1;
        if (kb + 1 < nkb) {
            issue_slab(kb + 1, buf ^ 1);
            // oldest 4 async ops (current slab) done; next slab stays in flight
            asm volatile("s_wait_asynccnt 0x4" ::: "memory");
        } else {
            asm volatile("s_wait_asynccnt 0x0" ::: "memory");
        }
        __syncthreads();

#pragma unroll
        for (int ks = 0; ks < 2; ++ks) {
            const int k0 = kb * 64 + ks * 32;
            __builtin_prefetch(pa + k0 + 256, 0, 1);   // global_prefetch_b8

            v16bf a;
            ((uint4*)&a)[0] = *(const uint4*)(pa + k0 + hi * 16);
            ((uint4*)&a)[1] = *(const uint4*)(pa + k0 + hi * 16 + 8);

#pragma unroll
            for (int j = 0; j < 4; ++j) {
                const bf16_t* lp = &sB[buf][j * 16 + lo][ks * 32 + hi * 16];
                v16bf b;
                ((uint4*)&b)[0] = ((const uint4*)lp)[0];
                ((uint4*)&b)[1] = ((const uint4*)lp)[1];
                acc[j] = __builtin_amdgcn_wmma_f32_16x16x32_bf16(
                    false, a, false, b, (short)0, acc[j], false, false);
            }
        }
        __syncthreads();   // all waves done reading buf before it is overwritten
    }

    // C fragment layout: element (m,n): m = row0 + r + 8*(lane/16), n = col0 + j*16 + lane%16
#pragma unroll
    for (int j = 0; j < 4; ++j) {
#pragma unroll
        for (int r = 0; r < 8; ++r) {
            int m = row0 + r + 8 * hi;
            int n = col0 + j * 16 + lo;
            C[(size_t)m * N + n] = acc[j][r];
        }
    }
}

// ---------------- RoPE + scatter to per-head layouts ----------------
__global__ __launch_bounds__(256)
void rope_scatter(const float* __restrict__ qkv, const float* __restrict__ cosb,
                  const float* __restrict__ sinb, bf16_t* __restrict__ qh,
                  bf16_t* __restrict__ kh, bf16_t* __restrict__ vt) {
    const int s = blockIdx.x;
    const float* row = qkv + (size_t)s * NQKV;
    for (int off = threadIdx.x; off < NQKV; off += 256) {
        const int d = off & 63;
        if (off < NH * HD) {                       // Q with RoPE
            const float c  = cosb[s * HD + d];
            const float sn = sinb[s * HD + d];
            const int h = off >> 6;
            const float x  = row[off];
            const float rh = (d < 32) ? -row[off + 32] : row[off - 32];
            qh[((size_t)h * S_LEN + s) * HD + d] = (bf16_t)(x * c + rh * sn);
        } else if (off < (NH + NKVH) * HD) {       // K with RoPE
            const float c  = cosb[s * HD + d];
            const float sn = sinb[s * HD + d];
            const int h = (off - NH * HD) >> 6;
            const float x  = row[off];
            const float rh = (d < 32) ? -row[off + 32] : row[off - 32];
            kh[((size_t)h * S_LEN + s) * HD + d] = (bf16_t)(x * c + rh * sn);
        } else {                                   // V, stored transposed [kvh][hd][s]
            const int h = (off - (NH + NKVH) * HD) >> 6;
            vt[((size_t)h * HD + d) * S_LEN + s] = (bf16_t)row[off];
        }
    }
}

// ---------------- causal flash attention, one wave per (16 rows x head) ------
__global__ __launch_bounds__(32)
void attn_fwd(const bf16_t* __restrict__ qh, const bf16_t* __restrict__ kh,
              const bf16_t* __restrict__ vt, bf16_t* __restrict__ attn) {
    __shared__ __align__(16) bf16_t sP[16 * 32];   // P tile, C-layout -> A-layout relay

    const int lane = threadIdx.x & 31;
    const int lo = lane & 15;
    const int hi = lane >> 4;
    const int qt = blockIdx.x;
    const int h  = blockIdx.y;
    const int kv = h >> 2;          // GROUPS = H / KVH = 4
    const int q0 = qt * 16;

    // Q A-fragments: HD=64 -> two K=32 chunks
    v16bf aq[2];
    {
        const bf16_t* qrow = qh + ((size_t)h * S_LEN + (q0 + lo)) * HD;
#pragma unroll
        for (int kk = 0; kk < 2; ++kk) {
            ((uint4*)&aq[kk])[0] = *(const uint4*)(qrow + kk * 32 + hi * 16);
            ((uint4*)&aq[kk])[1] = *(const uint4*)(qrow + kk * 32 + hi * 16 + 8);
        }
    }

    const v8f vzero = {0.f,0.f,0.f,0.f,0.f,0.f,0.f,0.f};
    v8f o[4] = {vzero, vzero, vzero, vzero};
    float rowmax[8], rowsum[8];
#pragma unroll
    for (int r = 0; r < 8; ++r) { rowmax[r] = -1e30f; rowsum[r] = 0.f; }

    const int ntiles = (q0 + 15) / 32 + 1;         // causal bound, key tiles of 32
    for (int kt = 0; kt < ntiles; ++kt) {
        // ---- scores S = Q K^T for 32 keys (two 16-col WMMA tiles, K=64 in 2 steps)
        v8f sc0 = vzero, sc1 = vzero;
#pragma unroll
        for (int kk = 0; kk < 2; ++kk) {
            const bf16_t* k0p =
                kh + ((size_t)kv * S_LEN + (kt * 32 + lo)) * HD + kk * 32 + hi * 16;
            const bf16_t* k1p = k0p + 16 * HD;
            v16bf b0, b1;
            ((uint4*)&b0)[0] = ((const uint4*)k0p)[0];
            ((uint4*)&b0)[1] = ((const uint4*)k0p)[1];
            ((uint4*)&b1)[0] = ((const uint4*)k1p)[0];
            ((uint4*)&b1)[1] = ((const uint4*)k1p)[1];
            sc0 = __builtin_amdgcn_wmma_f32_16x16x32_bf16(
                false, aq[kk], false, b0, (short)0, sc0, false, false);
            sc1 = __builtin_amdgcn_wmma_f32_16x16x32_bf16(
                false, aq[kk], false, b1, (short)0, sc1, false, false);
        }

        // ---- online softmax (fp32), row = r + 8*hi, cols spread over 16 lanes
        float fr[8];
#pragma unroll
        for (int r = 0; r < 8; ++r) {
            const int m  = q0 + r + 8 * hi;
            const int n0 = kt * 32 + lo;
            const int n1 = n0 + 16;
            float v0 = (n0 <= m) ? sc0[r] * ATT_SCALE : -1e30f;
            float v1 = (n1 <= m) ? sc1[r] * ATT_SCALE : -1e30f;
            float t = fmaxf(v0, v1);
#pragma unroll
            for (int off = 8; off >= 1; off >>= 1)
                t = fmaxf(t, __shfl_xor(t, off, 32));
            const float nm = fmaxf(rowmax[r], t);
            const float e0 = __expf(v0 - nm);
            const float e1 = __expf(v1 - nm);
            float sum = e0 + e1;
#pragma unroll
            for (int off = 8; off >= 1; off >>= 1)
                sum += __shfl_xor(sum, off, 32);
            const float f = __expf(rowmax[r] - nm);
            rowsum[r] = rowsum[r] * f + sum;
            rowmax[r] = nm;
            fr[r] = f;
            sP[(r + 8 * hi) * 32 + lo]      = (bf16_t)e0;
            sP[(r + 8 * hi) * 32 + 16 + lo] = (bf16_t)e1;
        }
#pragma unroll
        for (int j = 0; j < 4; ++j)
#pragma unroll
            for (int r = 0; r < 8; ++r) o[j][r] *= fr[r];

        __syncthreads();   // single wave: DS ops are in-order anyway

        // ---- reload P as an A-fragment (16x32 bf16) from LDS
        v16bf ap;
        ((uint4*)&ap)[0] = *(const uint4*)(sP + lo * 32 + hi * 16);
        ((uint4*)&ap)[1] = *(const uint4*)(sP + lo * 32 + hi * 16 + 8);

        // ---- O += P * V ; V transposed [kv][hd][s] makes B-fragments contiguous
#pragma unroll
        for (int j = 0; j < 4; ++j) {
            const bf16_t* vp =
                vt + ((size_t)kv * HD + (j * 16 + lo)) * S_LEN + kt * 32 + hi * 16;
            v16bf bv;
            ((uint4*)&bv)[0] = ((const uint4*)vp)[0];
            ((uint4*)&bv)[1] = ((const uint4*)vp)[1];
            o[j] = __builtin_amdgcn_wmma_f32_16x16x32_bf16(
                false, ap, false, bv, (short)0, o[j], false, false);
        }
        __syncthreads();
    }

    // ---- normalize and store [s][h*64 + d] as bf16 for the output projection
#pragma unroll
    for (int j = 0; j < 4; ++j)
#pragma unroll
        for (int r = 0; r < 8; ++r) {
            const int m = q0 + r + 8 * hi;
            attn[(size_t)m * (NH * HD) + h * HD + j * 16 + lo] =
                (bf16_t)(o[j][r] / rowsum[r]);
        }
}

// ---------------- host side ----------------
extern "C" void kernel_launch(void* const* d_in, const int* in_sizes, int n_in,
                              void* d_out, int out_size, void* d_ws, size_t ws_size,
                              hipStream_t stream) {
    const float* hid  = (const float*)d_in[0];
    const float* cosb = (const float*)d_in[1];
    const float* sinb = (const float*)d_in[2];
    const float* wq   = (const float*)d_in[3];
    const float* wk   = (const float*)d_in[4];
    const float* wv   = (const float*)d_in[5];
    const float* wo   = (const float*)d_in[6];
    float* out = (float*)d_out;

    char* ws = (char*)d_ws;
    bf16_t* hidb  = (bf16_t*)(ws + OFF_HID);
    bf16_t* wcat  = (bf16_t*)(ws + OFF_WCAT);
    bf16_t* wob   = (bf16_t*)(ws + OFF_WO);
    float*  qkv   = (float*)(ws + OFF_QKV);
    bf16_t* attnb = (bf16_t*)(ws + OFF_QKV);   // alias: qkv consumed before attn writes
    bf16_t* qh    = (bf16_t*)(ws + OFF_QH);
    bf16_t* khp   = (bf16_t*)(ws + OFF_KH);
    bf16_t* vtp   = (bf16_t*)(ws + OFF_VT);

    // 1) fp32 -> bf16 casts (weights fused into one [3072][2048] NT operand)
    {
        int n;
        n = S_LEN * D_DIM;
        cast_f32_bf16<<<(n + 255) / 256, 256, 0, stream>>>(hid, hidb, n);
        n = NH * HD * D_DIM;
        cast_f32_bf16<<<(n + 255) / 256, 256, 0, stream>>>(wq, wcat, n);
        n = NKVH * HD * D_DIM;
        cast_f32_bf16<<<(n + 255) / 256, 256, 0, stream>>>(
            wk, wcat + (size_t)NH * HD * D_DIM, n);
        cast_f32_bf16<<<(n + 255) / 256, 256, 0, stream>>>(
            wv, wcat + (size_t)(NH + NKVH) * HD * D_DIM, n);
        n = D_DIM * NH * HD;
        cast_f32_bf16<<<(n + 255) / 256, 256, 0, stream>>>(wo, wob, n);
    }

    // 2) fused QKV projection: [2048,3072] = hidden @ [wq|wk|wv]^T
    gemm_nt_wmma<<<dim3(S_LEN / 64, NQKV / 64), 128, 0, stream>>>(
        hidb, wcat, qkv, S_LEN, NQKV, D_DIM);

    // 3) RoPE + scatter to per-head Q/K and transposed V
    rope_scatter<<<S_LEN, 256, 0, stream>>>(qkv, cosb, sinb, qh, khp, vtp);

    // 4) causal flash attention (one wave per 16 query rows per head)
    attn_fwd<<<dim3(S_LEN / 16, NH), 32, 0, stream>>>(qh, khp, vtp, attnb);

    // 5) output projection: out = attn @ wo^T (fp32 result)
    gemm_nt_wmma<<<dim3(S_LEN / 64, D_DIM / 64), 128, 0, stream>>>(
        attnb, wob, out, S_LEN, D_DIM, NH * HD);
}
